// Attention_54546084659966
// MI455X (gfx1250) — matrix operand order repaired and checked by reading, not compiled, not run on hardware
//
#include <hip/hip_runtime.h>
#include <hip/hip_bf16.h>

typedef __bf16 bf16_t;
typedef __attribute__((ext_vector_type(16))) __bf16 v16bf;
typedef __attribute__((ext_vector_type(8)))  __bf16 v8bf;
typedef __attribute__((ext_vector_type(8)))  float  v8f;
typedef __attribute__((ext_vector_type(4)))  float  v4f;

#define NUM_HEADS 16
#define HEAD_DIM  64
#define DIM       1024
#define H_SELF    10
#define BATCH     16
#define SEQ       577
#define NPAD      608            // 19 * 32 key blocks
#define ROWS      (BATCH * SEQ)  // 9232
#define SCALEQ    0.125f         // 64^-0.5

// ---------------------------------------------------------------------------
// Async global->LDS copy (gfx1250 GLOBAL_LOAD_ASYNC_TO_LDS_B128, ASYNCcnt).
// ROCm builtin prototype (probe-confirmed): (global int4*, lds int4*, imm, imm).
// Guarded so a toolchain without the builtin falls back to sync LDS staging.
// ---------------------------------------------------------------------------
#if defined(__gfx1250__) && \
    __has_builtin(__builtin_amdgcn_global_load_async_to_lds_b128) && \
    __has_builtin(__builtin_amdgcn_s_wait_asynccnt)
#define HAVE_ASYNC_LDS 1
#else
#define HAVE_ASYNC_LDS 0
#endif

typedef int v4i_ __attribute__((vector_size(16)));
typedef __attribute__((address_space(1))) v4i_ g_v4i;
typedef __attribute__((address_space(3))) v4i_ l_v4i;

__device__ inline void async_cp16(bf16_t* lds_dst, const bf16_t* gsrc) {
#if HAVE_ASYNC_LDS
    __builtin_amdgcn_global_load_async_to_lds_b128(
        (g_v4i*)gsrc, (l_v4i*)lds_dst, 0, 0);
#else
    *(v8bf*)lds_dst = *(const v8bf*)gsrc;
#endif
}

__device__ inline void wait_async_lds() {
#if HAVE_ASYNC_LDS
    __builtin_amdgcn_s_wait_asynccnt(0);
#endif
}

// ---------------------------------------------------------------------------
// Fragment loaders following CDNA5 wave32 WMMA VGPR layouts (ISA 7.12.2).
// A (16x32 bf16): lane L holds row M=L&15; lanes 0-15: K=0..7,16..23;
//                 lanes 16-31: K=8..15,24..31.  Tile layout: [M][K], 16B aligned.
// B (32x16 bf16): lane L holds col N=L&15; lanes 0-15: K=0..7,16..23; 16-31: K=8..15,24..31.
//                 Tile layout: [N][K] (i.e. B transposed), 16B aligned.
// ---------------------------------------------------------------------------
__device__ inline v8bf ld8(const bf16_t* p) { return *(const v8bf*)p; }

__device__ inline v16bf frag_ld_a(const bf16_t* base, int stride) {
    const int lane = threadIdx.x & 31;
    const bf16_t* p = base + (lane & 15) * stride + ((lane >> 4) << 3);
    v8bf lo = ld8(p);
    v8bf hi = ld8(p + 16);
    return __builtin_shufflevector(lo, hi, 0,1,2,3,4,5,6,7,8,9,10,11,12,13,14,15);
}

__device__ inline v16bf frag_ld_b(const bf16_t* base, int stride) {
    const int lane = threadIdx.x & 31;
    const bf16_t* p = base + (lane & 15) * stride + ((lane >> 4) << 3);
    v8bf lo = ld8(p);
    v8bf hi = ld8(p + 16);
    return __builtin_shufflevector(lo, hi, 0,1,2,3,4,5,6,7,8,9,10,11,12,13,14,15);
}

__device__ inline v8f wmma_bf16(v16bf a, v16bf b, v8f c) {
    return __builtin_amdgcn_wmma_f32_16x16x32_bf16(false, a, false, b, (short)0, c,
                                                   false, false);
}

// ---------------------------------------------------------------------------
// Weight convert+transpose: W[IN][OUT] (f32) -> WT[OUT][IN] (bf16)
// ---------------------------------------------------------------------------
__global__ void k_transpose_cvt(const float* __restrict__ W, bf16_t* __restrict__ WT,
                                int IN, int OUT) {
    int idx = blockIdx.x * blockDim.x + threadIdx.x;
    int total = IN * OUT;
    if (idx >= total) return;
    int o = idx / IN;
    int i = idx - o * IN;
    WT[idx] = (bf16_t)W[(size_t)i * OUT + o];
}

// ---------------------------------------------------------------------------
// QKV GEMM: x[s] [9232][1024] f32 @ WqkvT [3072][1024] bf16 -> q/k/v bf16.
// Block = 4 waves, 64x64 tile; each wave owns 32x32 (2x2 fragments, 4 WMMA/step).
// Double-buffered LDS, one barrier per 32-deep k-step. Q pre-scaled by d^-1/2;
// V stored transposed [b,h,d,n] so attention's P*V B-fragments are contiguous.
// ---------------------------------------------------------------------------
__global__ __launch_bounds__(128)
void k_qkv_gemm(const float* __restrict__ x1, const float* __restrict__ x2,
                const bf16_t* __restrict__ WT,
                bf16_t* __restrict__ qb, bf16_t* __restrict__ kbuf,
                bf16_t* __restrict__ vb) {
    const int s    = blockIdx.z;
    const float* x = s ? x2 : x1;
    const int rt   = blockIdx.y;        // 64-row tile
    const int cb   = blockIdx.x * 64;   // col base
    const int wave = threadIdx.x >> 5;
    const int lane = threadIdx.x & 31;
    const int t    = threadIdx.x;
    const int wr   = (wave >> 1) * 32;  // wave row offset in tile
    const int wc   = (wave & 1) * 32;   // wave col offset in tile

    __shared__ bf16_t sA[2][64 * 40];
    __shared__ bf16_t sB[2][64 * 40];

    // two-phase stage: issue all 4 global loads, then convert+store — lets the
    // compiler stagger s_wait_loadcnt (0x3..0x0) instead of full-wait per chunk
    auto stageA = [&](int k0, int bi) {
        v4f f[4];
#pragma unroll
        for (int i = 0; i < 4; ++i) {       // 512 float4 chunks, 4 per thread
            int id = t + i * 128;
            int r  = id >> 3;
            int ch = (id & 7) << 2;
            int rg = rt * 64 + r;
            if (rg > ROWS - 1) rg = ROWS - 1;   // clamp: no exec-branch in loop
            f[i] = *(const v4f*)(x + (size_t)rg * DIM + k0 + ch);
        }
#pragma unroll
        for (int i = 0; i < 4; ++i) {
            int id = t + i * 128;
            int r  = id >> 3;
            int ch = (id & 7) << 2;
            bf16_t* d = sA[bi] + r * 40 + ch;
            d[0] = (bf16_t)f[i].x; d[1] = (bf16_t)f[i].y;
            d[2] = (bf16_t)f[i].z; d[3] = (bf16_t)f[i].w;
        }
    };
    auto stageB = [&](int k0, int bi) {
#pragma unroll
        for (int i = 0; i < 2; ++i) {       // 256 16B chunks, async
            int id = t + i * 128;
            int r  = id >> 2;
            int ch = (id & 3) << 3;
            async_cp16(sB[bi] + r * 40 + ch, WT + (size_t)(cb + r) * DIM + k0 + ch);
        }
    };

    v8f acc[2][2];
#pragma unroll
    for (int i = 0; i < 2; ++i)
#pragma unroll
        for (int j = 0; j < 2; ++j)
#pragma unroll
            for (int v = 0; v < 8; ++v) acc[i][j][v] = 0.0f;

    stageA(0, 0);
    stageB(0, 0);
    int buf = 0;
    for (int k0 = 0; k0 < DIM; k0 += 32) {
        wait_async_lds();
        __syncthreads();
        const bf16_t* A = sA[buf];
        const bf16_t* B = sB[buf];
        v16bf a0 = frag_ld_a(A + (wr + 0) * 40, 40);
        v16bf a1 = frag_ld_a(A + (wr + 16) * 40, 40);
        v16bf b0 = frag_ld_b(B + (wc + 0) * 40, 40);
        v16bf b1 = frag_ld_b(B + (wc + 16) * 40, 40);
        if (k0 + 32 < DIM) { stageA(k0 + 32, buf ^ 1); stageB(k0 + 32, buf ^ 1); }
        acc[0][0] = wmma_bf16(a0, b0, acc[0][0]);
        acc[0][1] = wmma_bf16(a0, b1, acc[0][1]);
        acc[1][0] = wmma_bf16(a1, b0, acc[1][0]);
        acc[1][1] = wmma_bf16(a1, b1, acc[1][1]);
        buf ^= 1;
    }

    const int half8 = (lane >> 4) << 3;
#pragma unroll
    for (int j = 0; j < 2; ++j) {
        const int col   = cb + wc + j * 16 + (lane & 15);
        const int which = col >> 10;        // 0=q 1=k 2=v (uniform per wave,j)
        const int rem   = col & 1023;
        const int h     = rem >> 6;
        const int d     = rem & 63;
#pragma unroll
        for (int i = 0; i < 2; ++i) {
#pragma unroll
            for (int v = 0; v < 8; ++v) {
                int rg = rt * 64 + wr + i * 16 + half8 + v;
                if (rg >= ROWS) continue;
                int bi = rg / SEQ;
                int n  = rg - bi * SEQ;
                size_t bh = (size_t)(s * BATCH + bi) * NUM_HEADS + h;
                float val = acc[i][j][v];
                if (which == 0)
                    qb[(bh * NPAD + n) * HEAD_DIM + d] = (bf16_t)(val * SCALEQ);
                else if (which == 1)
                    kbuf[(bh * NPAD + n) * HEAD_DIM + d] = (bf16_t)val;
                else
                    vb[(bh * HEAD_DIM + d) * NPAD + n] = (bf16_t)val;
            }
        }
    }
}

// ---------------------------------------------------------------------------
// Flash attention: one wave per 16-query tile; 4 waves per block share
// double-buffered, async-staged K/V 32-key LDS tiles.
// KV stream = own stream for h<10 (self), other stream for h>=10 (cross).
// ---------------------------------------------------------------------------
__global__ __launch_bounds__(128)
void k_attn(const bf16_t* __restrict__ qb, const bf16_t* __restrict__ kbuf,
            const bf16_t* __restrict__ vb, bf16_t* __restrict__ ab) {
    const int wave = threadIdx.x >> 5;
    const int lane = threadIdx.x & 31;
    const int t    = threadIdx.x;
    const int h    = blockIdx.y;
    const int s    = blockIdx.z >> 4;
    const int b    = blockIdx.z & 15;
    int qt = blockIdx.x * 4 + wave;
    if (qt > 36) qt = 36;              // duplicated tail work stores identical data

    const int kvs = (h < H_SELF) ? s : (1 - s);
    const size_t bhq = (size_t)(s * BATCH + b) * NUM_HEADS + h;
    const size_t bhk = (size_t)(kvs * BATCH + b) * NUM_HEADS + h;

    __shared__ bf16_t sK[2][32 * 80];       // [key][d]
    __shared__ bf16_t sV[2][64 * 48];       // [d][key]  (V transposed)
    __shared__ bf16_t sP[4][16 * 40];       // per-wave P tile [m][key]

    const bf16_t* qptr  = qb + (bhq * NPAD + (size_t)qt * 16) * HEAD_DIM;
    const bf16_t* kbase = kbuf + bhk * (size_t)NPAD * HEAD_DIM;
    const bf16_t* vbase = vb + bhk * (size_t)HEAD_DIM * NPAD;

    auto stageKV = [&](int blk, int bi) {
        const int kb0 = blk * 32;
#pragma unroll
        for (int i = 0; i < 2; ++i) {       // K: 32 keys x 64 d
            int id = t + i * 128;
            int r  = id >> 3;
            int ch = (id & 7) << 3;
            async_cp16(sK[bi] + r * 80 + ch,
                       kbase + (size_t)(kb0 + r) * HEAD_DIM + ch);
        }
#pragma unroll
        for (int i = 0; i < 2; ++i) {       // V: 64 d x 32 keys
            int id = t + i * 128;
            int r  = id >> 2;
            int ch = (id & 3) << 3;
            async_cp16(sV[bi] + r * 48 + ch,
                       vbase + (size_t)r * NPAD + kb0 + ch);
        }
    };

    v16bf qf0 = frag_ld_a(qptr, HEAD_DIM);        // K = 0..31
    v16bf qf1 = frag_ld_a(qptr + 32, HEAD_DIM);   // K = 32..63

    v8f o[4];
    float m_[8], l_[8];
#pragma unroll
    for (int v = 0; v < 8; ++v) { m_[v] = -3.0e38f; l_[v] = 0.0f; }
#pragma unroll
    for (int c = 0; c < 4; ++c)
#pragma unroll
        for (int v = 0; v < 8; ++v) o[c][v] = 0.0f;

    v8f z;
#pragma unroll
    for (int v = 0; v < 8; ++v) z[v] = 0.0f;

    stageKV(0, 0);
    int buf = 0;
    for (int blk = 0; blk < NPAD / 32; ++blk) {
        const int kb0 = blk * 32;
        wait_async_lds();
        __syncthreads();                       // current K/V tile visible
        if (blk + 1 < NPAD / 32) stageKV(blk + 1, buf ^ 1);

        const bf16_t* K = sK[buf];
        const bf16_t* V = sV[buf];

        // S = Q @ K^T  (two 16-key subtiles, K-dim 64 = 2 WMMAs each)
        v8f s0 = wmma_bf16(qf0, frag_ld_b(K, 80), z);
        s0 = wmma_bf16(qf1, frag_ld_b(K + 32, 80), s0);
        v8f s1 = wmma_bf16(qf0, frag_ld_b(K + 16 * 80, 80), z);
        s1 = wmma_bf16(qf1, frag_ld_b(K + 16 * 80 + 32, 80), s1);

        const int n0 = kb0 + (lane & 15);
        const int n1 = n0 + 16;
        bf16_t* pw = sP[wave];
#pragma unroll
        for (int v = 0; v < 8; ++v) {
            float a0 = (n0 < SEQ) ? s0[v] : -3.0e38f;
            float a1 = (n1 < SEQ) ? s1[v] : -3.0e38f;
            float mx = fmaxf(a0, a1);
            mx = fmaxf(mx, __shfl_xor(mx, 1));
            mx = fmaxf(mx, __shfl_xor(mx, 2));
            mx = fmaxf(mx, __shfl_xor(mx, 4));
            mx = fmaxf(mx, __shfl_xor(mx, 8));   // row max across 16-lane half
            float mnew  = fmaxf(m_[v], mx);
            float alpha = __expf(m_[v] - mnew);
            float p0 = __expf(a0 - mnew);
            float p1 = __expf(a1 - mnew);
            float rs = p0 + p1;
            rs += __shfl_xor(rs, 1);
            rs += __shfl_xor(rs, 2);
            rs += __shfl_xor(rs, 4);
            rs += __shfl_xor(rs, 8);
            l_[v] = l_[v] * alpha + rs;
            m_[v] = mnew;
            o[0][v] *= alpha; o[1][v] *= alpha;
            o[2][v] *= alpha; o[3][v] *= alpha;
            int mrow = v + ((lane >> 4) << 3);
            pw[mrow * 40 + (lane & 15)]      = (bf16_t)p0;
            pw[mrow * 40 + 16 + (lane & 15)] = (bf16_t)p1;
        }
        __syncthreads();                       // P visible for fragment reload
        v16bf pf = frag_ld_a(pw, 40);
#pragma unroll
        for (int c = 0; c < 4; ++c)
            o[c] = wmma_bf16(pf, frag_ld_b(V + c * 16 * 48, 48), o[c]);
        buf ^= 1;
    }

    // epilogue: O / l -> concat layout [s][b][n][h*64+d] in bf16
#pragma unroll
    for (int v = 0; v < 8; ++v) {
        int mrow = v + ((lane >> 4) << 3);
        int n = qt * 16 + mrow;
        if (n >= SEQ) continue;
        float inv = 1.0f / l_[v];
        size_t rowoff = (((size_t)s * BATCH + b) * SEQ + n) * DIM + h * HEAD_DIM;
#pragma unroll
        for (int c = 0; c < 4; ++c)
            ab[rowoff + c * 16 + (lane & 15)] = (bf16_t)(o[c][v] * inv);
    }
}

// ---------------------------------------------------------------------------
// Output projection: ab[s] [9232][1024] bf16 @ WprojT [1024][1024] bf16 + bias.
// Same 64x64 double-buffered tiling; both tiles async-staged.
// ---------------------------------------------------------------------------
__global__ __launch_bounds__(128)
void k_proj(const bf16_t* __restrict__ ab, const bf16_t* __restrict__ WT,
            const float* __restrict__ bias, float* __restrict__ out) {
    const int s    = blockIdx.z;
    const int rt   = blockIdx.y;
    const int cb   = blockIdx.x * 64;
    const int wave = threadIdx.x >> 5;
    const int lane = threadIdx.x & 31;
    const int t    = threadIdx.x;
    const int wr   = (wave >> 1) * 32;
    const int wc   = (wave & 1) * 32;

    __shared__ bf16_t sA[2][64 * 40];
    __shared__ bf16_t sB[2][64 * 40];

    const bf16_t* A = ab + (size_t)s * ROWS * DIM;

    auto stageA = [&](int k0, int bi) {
#pragma unroll
        for (int i = 0; i < 2; ++i) {
            int id = t + i * 128;
            int r  = id >> 2;
            int ch = (id & 3) << 3;
            int rg = rt * 64 + r;
            if (rg > ROWS - 1) rg = ROWS - 1;
            async_cp16(sA[bi] + r * 40 + ch, A + (size_t)rg * DIM + k0 + ch);
        }
    };
    auto stageB = [&](int k0, int bi) {
#pragma unroll
        for (int i = 0; i < 2; ++i) {
            int id = t + i * 128;
            int r  = id >> 2;
            int ch = (id & 3) << 3;
            async_cp16(sB[bi] + r * 40 + ch, WT + (size_t)(cb + r) * DIM + k0 + ch);
        }
    };

    v8f acc[2][2];
#pragma unroll
    for (int i = 0; i < 2; ++i)
#pragma unroll
        for (int j = 0; j < 2; ++j)
#pragma unroll
            for (int v = 0; v < 8; ++v) acc[i][j][v] = 0.0f;

    stageA(0, 0);
    stageB(0, 0);
    int buf = 0;
    for (int k0 = 0; k0 < DIM; k0 += 32) {
        wait_async_lds();
        __syncthreads();
        const bf16_t* At = sA[buf];
        const bf16_t* Bt = sB[buf];
        v16bf a0 = frag_ld_a(At + (wr + 0) * 40, 40);
        v16bf a1 = frag_ld_a(At + (wr + 16) * 40, 40);
        v16bf b0 = frag_ld_b(Bt + (wc + 0) * 40, 40);
        v16bf b1 = frag_ld_b(Bt + (wc + 16) * 40, 40);
        if (k0 + 32 < DIM) { stageA(k0 + 32, buf ^ 1); stageB(k0 + 32, buf ^ 1); }
        acc[0][0] = wmma_bf16(a0, b0, acc[0][0]);
        acc[0][1] = wmma_bf16(a0, b1, acc[0][1]);
        acc[1][0] = wmma_bf16(a1, b0, acc[1][0]);
        acc[1][1] = wmma_bf16(a1, b1, acc[1][1]);
        buf ^= 1;
    }

    const int half8 = (lane >> 4) << 3;
#pragma unroll
    for (int j = 0; j < 2; ++j) {
        const int col  = cb + wc + j * 16 + (lane & 15);
        const float bv = bias[col];
#pragma unroll
        for (int i = 0; i < 2; ++i) {
#pragma unroll
            for (int v = 0; v < 8; ++v) {
                int rg = rt * 64 + wr + i * 16 + half8 + v;
                if (rg >= ROWS) continue;
                out[((size_t)s * ROWS + rg) * DIM + col] = acc[i][j][v] + bv;
            }
        }
    }
}

// ---------------------------------------------------------------------------
extern "C" void kernel_launch(void* const* d_in, const int* in_sizes, int n_in,
                              void* d_out, int out_size, void* d_ws, size_t ws_size,
                              hipStream_t stream) {
    (void)in_sizes; (void)n_in; (void)out_size; (void)ws_size;
    const float* x1    = (const float*)d_in[0];
    const float* x2    = (const float*)d_in[1];
    const float* Wqkv  = (const float*)d_in[2];
    const float* Wproj = (const float*)d_in[3];
    const float* bproj = (const float*)d_in[4];
    float* out = (float*)d_out;

    char* ws = (char*)d_ws;
    size_t off = 0;
    auto carve = [&](size_t bytes) -> char* {
        char* p = ws + off;
        off = (off + bytes + 255) & ~(size_t)255;
        return p;
    };
    bf16_t* wqkvT  = (bf16_t*)carve((size_t)3072 * 1024 * 2);
    bf16_t* wprojT = (bf16_t*)carve((size_t)1024 * 1024 * 2);
    const size_t qkvBytes = (size_t)2 * BATCH * NUM_HEADS * NPAD * HEAD_DIM * 2;
    bf16_t* qb = (bf16_t*)carve(qkvBytes);
    bf16_t* kb = (bf16_t*)carve(qkvBytes);
    bf16_t* vb = (bf16_t*)carve(qkvBytes);
    bf16_t* ab = (bf16_t*)carve((size_t)2 * ROWS * DIM * 2);

    {
        int total = 3072 * 1024;
        k_transpose_cvt<<<(total + 255) / 256, 256, 0, stream>>>(Wqkv, wqkvT, 1024, 3072);
    }
    {
        int total = 1024 * 1024;
        k_transpose_cvt<<<(total + 255) / 256, 256, 0, stream>>>(Wproj, wprojT, 1024, 1024);
    }
    {
        dim3 g(48, (ROWS + 63) / 64, 2);   // 48 col-blocks x 145 row-tiles x 2
        k_qkv_gemm<<<g, 128, 0, stream>>>(x1, x2, wqkvT, qb, kb, vb);
    }
    {
        dim3 g(10, NUM_HEADS, 32);         // 10 qtile-groups x 16 heads x (2s*16b)
        k_attn<<<g, 128, 0, stream>>>(qb, kb, vb, ab);
    }
    {
        dim3 g(16, (ROWS + 63) / 64, 2);
        k_proj<<<g, 128, 0, stream>>>(ab, wprojT, bproj, out);
    }
}
